// GatedLinearRecurrence_37374805409902
// MI455X (gfx1250) — matrix operand order repaired and checked by reading, not compiled, not run on hardware
//
#include <hip/hip_runtime.h>
#include <hip/hip_bf16.h>
#include <math.h>

// ---------------------------------------------------------------------------
// RG-LRU (Griffin) fused pipeline for MI455X / gfx1250, wave32 + WMMA bf16.
//   B=4, S=4096, D_IN=D_ST=1024.
// Workspace layout (bytes):
//   [0      , 2MB)   W_in  bf16 [1024][1024]
//   [2MB    , 4MB)   W_r   bf16
//   [4MB    , 6MB)   W_i   bf16
//   [6MB    , 8MB)   W_out bf16
//   [8MB    , +4KB)  c[n] = -8*softplus(log_decay[n])  (fp32)
//   [16MB   , 80MB)  a  fp32 [16384][1024]
//   [80MB   ,144MB)  u  fp32 [16384][1024]
//   [144MB  ,176MB)  hs bf16 [16384][1024]
//   [176MB  ,208MB)  x  bf16 [16384][1024]
// ---------------------------------------------------------------------------

typedef __attribute__((ext_vector_type(16))) __bf16 v16bf;
typedef __attribute__((ext_vector_type(8)))  __bf16 v8bf;
typedef __attribute__((ext_vector_type(4)))  __bf16 v4bf;
typedef __attribute__((ext_vector_type(8)))  float  v8f;
typedef __attribute__((ext_vector_type(4)))  float  v4f;

#define NB   4
#define NS   4096
#define NDIN 1024
#define NDST 1024
#define NM   (NB * NS)      // 16384 rows

// gfx1250 async global->LDS copy path (ASYNCcnt), with safe fallback.
#if defined(__has_builtin)
#  if __has_builtin(__builtin_amdgcn_global_load_async_to_lds_b128) && \
      __has_builtin(__builtin_amdgcn_s_wait_asynccnt)
#    define USE_ASYNC_LDS 1
#  endif
#endif
#ifndef USE_ASYNC_LDS
#  define USE_ASYNC_LDS 0
#endif

union ABf { v16bf v; v8bf h[2]; };

__device__ __forceinline__ float sigmoidf_(float x) {
  return 1.0f / (1.0f + __expf(-x));
}

#if USE_ASYNC_LDS
typedef int v4i32 __attribute__((vector_size(16)));
typedef __attribute__((address_space(1))) v4i32 as1_v4i32;
typedef __attribute__((address_space(3))) v4i32 as3_v4i32;
#endif

__device__ __forceinline__ void copy16_to_lds(const __bf16* gsrc, __bf16* ldst) {
#if USE_ASYNC_LDS
  // flat->AS1 is a no-op bitcast; flat->AS3 is a trunc-to-32 (LLVM lowering),
  // so integer round-trips are exact.
  __builtin_amdgcn_global_load_async_to_lds_b128(
      (as1_v4i32*)(uintptr_t)gsrc,
      (as3_v4i32*)(uint32_t)(uintptr_t)ldst, 0, 0);
#else
  *(v8bf*)ldst = *(const v8bf*)gsrc;
#endif
}

// Each stage issues 2 async copy instructions per wave; with one stage in
// flight behind the current one, wait for cnt<=2, else drain to 0.
__device__ __forceinline__ void wait_stage(int more_in_flight) {
#if USE_ASYNC_LDS
  if (more_in_flight) __builtin_amdgcn_s_wait_asynccnt(2);
  else                __builtin_amdgcn_s_wait_asynccnt(0);
#else
  (void)more_in_flight;
#endif
}

// ------------------------- fp32 -> bf16 conversion -------------------------
__global__ void __launch_bounds__(256)
k_f2bf(const float* __restrict__ src, __bf16* __restrict__ dst) {
  size_t i4 = ((size_t)blockIdx.x * 256 + threadIdx.x) * 4;
  v4f v = *(const v4f*)(src + i4);
  v4bf o;
  o[0] = (__bf16)v[0]; o[1] = (__bf16)v[1];
  o[2] = (__bf16)v[2]; o[3] = (__bf16)v[3];
  *(v4bf*)(dst + i4) = o;
}

// ------------------------- decay coefficients ------------------------------
__global__ void __launch_bounds__(256)
k_decay(const float* __restrict__ log_decay, float* __restrict__ c) {
  int n = blockIdx.x * 256 + threadIdx.x;
  float x = log_decay[n];
  float sp = fmaxf(x, 0.0f) + log1pf(__expf(-fabsf(x)));   // stable softplus
  c[n] = -8.0f * sp;
}

// ------------------------- fused gates GEMM --------------------------------
// Workgroup tile: M=64, N=128. 8 waves as 2(M) x 4(N); each wave owns a
// 32x32 output tile for all three heads (p, r, i): 12 accumulators.
// Every B fragment is reused by 2 M-subtiles -> 12 WMMAs per 12 B-loads
// per k-substep (~32 FLOP/byte of B traffic).
// x (bf16) staged to double-buffered LDS via async copies, K_STAGE=64.
__global__ void __launch_bounds__(256)
k_gates(const __bf16* __restrict__ xbf, const __bf16* __restrict__ Wbf,
        const float* __restrict__ b_in, const float* __restrict__ b_r,
        const float* __restrict__ b_i, const float* __restrict__ cdec,
        float* __restrict__ a_ws, float* __restrict__ u_ws) {
  __shared__ __bf16 lx[2][64 * 72];                // pitch 72 bf16 (144B)

  const int tid  = threadIdx.x;
  const int wave = tid >> 5, lane = tid & 31;
  const int half = lane >> 4, l16 = lane & 15;
  const int m_blk = blockIdx.x * 64;
  const int wm = wave & 1, wn = wave >> 1;
  const int m0 = m_blk + wm * 32;
  const int n0 = blockIdx.y * 128 + wn * 32;

  // staging: 512 16B-chunks (64 rows x 8), 2 per thread
  const int r_st0 = tid >> 3;                      // rows 0..31
  const int c_st  = (tid & 7) * 8;                 // col (x8 bf16 = 16B)

  const v8f vzero = {0.f,0.f,0.f,0.f,0.f,0.f,0.f,0.f};
  v8f acc[3][2][2];                                // [head][ncol][msub]
#pragma unroll
  for (int hd = 0; hd < 3; ++hd)
#pragma unroll
    for (int nc = 0; nc < 2; ++nc)
#pragma unroll
      for (int ms = 0; ms < 2; ++ms) acc[hd][nc][ms] = vzero;

  auto stage = [&](int buf, int kk) {
    copy16_to_lds(xbf + (size_t)(m_blk + r_st0) * NDIN + kk + c_st,
                  &lx[buf][r_st0 * 72 + c_st]);
    copy16_to_lds(xbf + (size_t)(m_blk + r_st0 + 32) * NDIN + kk + c_st,
                  &lx[buf][(r_st0 + 32) * 72 + c_st]);
  };
  auto loadB = [&](ABf (&Bf)[3][2], int k) {
#pragma unroll
    for (int hd = 0; hd < 3; ++hd)
#pragma unroll
      for (int nc = 0; nc < 2; ++nc) {
        // B (32x16): elem j -> K = 16*(j>>3) + (j&7) + 8*half, N = l16
        const __bf16* brow = Wbf + (size_t)hd * (NDST * NDIN) +
                             (size_t)(n0 + nc * 16 + l16) * NDIN + k + 8 * half;
        Bf[hd][nc].h[0] = *(const v8bf*)(brow);
        Bf[hd][nc].h[1] = *(const v8bf*)(brow + 16);
      }
  };

  ABf Bcur[3][2], Bnxt[3][2];
  loadB(Bcur, 0);
  stage(0, 0);

  for (int kk = 0; kk < NDIN; kk += 64) {
    const int buf = (kk >> 6) & 1;
    if (kk + 64 < NDIN) {
      stage(buf ^ 1, kk + 64);
      wait_stage(1);
    } else {
      wait_stage(0);
    }
    __syncthreads();

#pragma unroll
    for (int ks = 0; ks < 2; ++ks) {
      const int k = kk + ks * 32;
      // A (16x32): elem j -> K = 16*(j>>3) + (j&7) + 8*half, M = l16
      ABf A[2];
#pragma unroll
      for (int ms = 0; ms < 2; ++ms) {
        const __bf16* arow =
            &lx[buf][(wm * 32 + ms * 16 + l16) * 72 + ks * 32];
        A[ms].h[0] = *(const v8bf*)(arow + 8 * half);
        A[ms].h[1] = *(const v8bf*)(arow + 16 + 8 * half);
      }

      if (k + 32 < NDIN) loadB(Bnxt, k + 32);   // pipeline next substep

#pragma unroll
      for (int hd = 0; hd < 3; ++hd)
#pragma unroll
        for (int nc = 0; nc < 2; ++nc)
#pragma unroll
          for (int ms = 0; ms < 2; ++ms)
            acc[hd][nc][ms] = __builtin_amdgcn_wmma_f32_16x16x32_bf16(
                false, A[ms].v, false, Bcur[hd][nc].v, (short)0,
                acc[hd][nc][ms], false, false);

#pragma unroll
      for (int hd = 0; hd < 3; ++hd)
#pragma unroll
        for (int nc = 0; nc < 2; ++nc) Bcur[hd][nc] = Bnxt[hd][nc];
    }
    __syncthreads();   // protect buf from being restaged under slow waves
  }

  // epilogue: gates -> a, u
#pragma unroll
  for (int nc = 0; nc < 2; ++nc) {
    const int n = n0 + nc * 16 + l16;
    const float bi = b_in[n], br = b_r[n], bg = b_i[n], cd = cdec[n];
#pragma unroll
    for (int ms = 0; ms < 2; ++ms) {
#pragma unroll
      for (int rr = 0; rr < 8; ++rr) {
        const int m = m0 + ms * 16 + rr + 8 * half;  // C/D: M = r + 8*half
        float p  = acc[0][nc][ms][rr] + bi;
        float rg = sigmoidf_(acc[1][nc][ms][rr] + br);
        float ig = sigmoidf_(acc[2][nc][ms][rr] + bg);
        float ag = sigmoidf_(cd * rg);
        float sc = sqrtf(1.0f - ag * ag + 1e-6f);
        float ug = sc * ig * p;
        size_t idx = (size_t)m * NDST + n;
        a_ws[idx] = ag;
        u_ws[idx] = ug;
      }
    }
  }
}

// ------------------------- sequential scan ---------------------------------
// One thread per (b, n) channel: h = a*h + u over S steps. 4096 channels.
__global__ void __launch_bounds__(256)
k_scan(const float* __restrict__ a_ws, const float* __restrict__ u_ws,
       const float* __restrict__ h0, __bf16* __restrict__ hs,
       float* __restrict__ h_last) {
  const int b = blockIdx.x >> 2;
  const int n = ((blockIdx.x & 3) << 8) + threadIdx.x;
  const size_t base = (size_t)b * NS * NDST + n;
  float h = h0[b * NDST + n];
#pragma unroll 4
  for (int s = 0; s < NS; ++s) {
    size_t idx = base + (size_t)s * NDST;
    if (s + 16 < NS) {                              // gfx1250 global_prefetch
      __builtin_prefetch(a_ws + idx + 16 * NDST, 0, 0);
      __builtin_prefetch(u_ws + idx + 16 * NDST, 0, 0);
    }
    h = fmaf(a_ws[idx], h, u_ws[idx]);
    hs[idx] = (__bf16)h;
  }
  h_last[b * NDST + n] = h;
}

// ------------------------- output GEMM -------------------------------------
// out[m][d] = sum_n hs[m][n] * W_out[d][n] + b_out[d].
// Workgroup tile M=64, D=256; 8 waves as 2(M) x 4(D); wave = 32x64 (8 WMMA
// per k-step; B frags reused by 2 M-subtiles, A frags by 4 D-tiles).
// A and B fragments software-pipelined one k-step ahead.
__global__ void __launch_bounds__(256)
k_out(const __bf16* __restrict__ hs, const __bf16* __restrict__ Wobf,
      const float* __restrict__ b_out, float* __restrict__ out) {
  const int tid  = threadIdx.x;
  const int wave = tid >> 5, lane = tid & 31;
  const int half = lane >> 4, l16 = lane & 15;
  const int m0 = blockIdx.x * 64 + (wave & 1) * 32;
  const int d0 = blockIdx.y * 256 + (wave >> 1) * 64;

  const v8f vzero = {0.f,0.f,0.f,0.f,0.f,0.f,0.f,0.f};
  v8f acc[4][2];                                    // [dtile][msub]
#pragma unroll
  for (int t = 0; t < 4; ++t)
#pragma unroll
    for (int ms = 0; ms < 2; ++ms) acc[t][ms] = vzero;

  auto loadA = [&](ABf (&A)[2], int k) {
#pragma unroll
    for (int ms = 0; ms < 2; ++ms) {
      const __bf16* arow = hs + (size_t)(m0 + ms * 16 + l16) * NDST + k;
      A[ms].h[0] = *(const v8bf*)(arow + 8 * half);
      A[ms].h[1] = *(const v8bf*)(arow + 16 + 8 * half);
    }
  };
  auto loadB = [&](ABf (&Bf)[4], int k) {
#pragma unroll
    for (int t = 0; t < 4; ++t) {
      const __bf16* brow =
          Wobf + (size_t)(d0 + t * 16 + l16) * NDST + k + 8 * half;
      Bf[t].h[0] = *(const v8bf*)(brow);
      Bf[t].h[1] = *(const v8bf*)(brow + 16);
    }
  };

  ABf A0[2], A1[2], B0[4], B1[4];
  loadA(A0, 0);
  loadB(B0, 0);

  for (int kk = 0; kk < NDST; kk += 32) {
    if (kk + 32 < NDST) {             // pipeline next k-step
      loadA(A1, kk + 32);
      loadB(B1, kk + 32);
    }
#pragma unroll
    for (int t = 0; t < 4; ++t)
#pragma unroll
      for (int ms = 0; ms < 2; ++ms)
        acc[t][ms] = __builtin_amdgcn_wmma_f32_16x16x32_bf16(
            false, A0[ms].v, false, B0[t].v, (short)0, acc[t][ms],
            false, false);
#pragma unroll
    for (int ms = 0; ms < 2; ++ms) A0[ms] = A1[ms];
#pragma unroll
    for (int t = 0; t < 4; ++t) B0[t] = B1[t];
  }

#pragma unroll
  for (int t = 0; t < 4; ++t) {
    const int d = d0 + t * 16 + l16;
    const float bo = b_out[d];
#pragma unroll
    for (int ms = 0; ms < 2; ++ms) {
#pragma unroll
      for (int rr = 0; rr < 8; ++rr) {
        const int m = m0 + ms * 16 + rr + 8 * half;
        out[(size_t)m * NDIN + d] = acc[t][ms][rr] + bo;
      }
    }
  }
}

// ---------------------------------------------------------------------------
extern "C" void kernel_launch(void* const* d_in, const int* in_sizes, int n_in,
                              void* d_out, int out_size, void* d_ws,
                              size_t ws_size, hipStream_t stream) {
  (void)in_sizes; (void)n_in; (void)out_size; (void)ws_size;
  const float* x         = (const float*)d_in[0];
  const float* h0        = (const float*)d_in[1];
  const float* W_in      = (const float*)d_in[2];
  const float* b_in      = (const float*)d_in[3];
  const float* W_r       = (const float*)d_in[4];
  const float* b_r       = (const float*)d_in[5];
  const float* W_i       = (const float*)d_in[6];
  const float* b_i       = (const float*)d_in[7];
  const float* log_decay = (const float*)d_in[8];
  const float* W_out     = (const float*)d_in[9];
  const float* b_out     = (const float*)d_in[10];

  char* ws = (char*)d_ws;
  __bf16* Wbf   = (__bf16*)(ws);                        // 3 x 2MB (in,r,i)
  __bf16* Wobf  = (__bf16*)(ws + (3ull << 21));         // 2MB
  float*  cdec  = (float*)(ws + (4ull << 21));          // 4KB
  float*  a_ws  = (float*)(ws + (16ull << 20));         // 64MB
  float*  u_ws  = (float*)(ws + (80ull << 20));         // 64MB
  __bf16* hs    = (__bf16*)(ws + (144ull << 20));       // 32MB
  __bf16* xbf   = (__bf16*)(ws + (176ull << 20));       // 32MB

  float* out    = (float*)d_out;                        // (B,S,D_IN) fp32
  float* h_last = out + (size_t)NM * NDIN;              // (B,D_ST)   fp32

  const int wgrid = (NDST * NDIN) / (256 * 4);          // 1024 blocks / matrix
  k_f2bf<<<dim3(wgrid), 256, 0, stream>>>(W_in,  Wbf);
  k_f2bf<<<dim3(wgrid), 256, 0, stream>>>(W_r,   Wbf + (size_t)NDST * NDIN);
  k_f2bf<<<dim3(wgrid), 256, 0, stream>>>(W_i,   Wbf + 2ull * NDST * NDIN);
  k_f2bf<<<dim3(wgrid), 256, 0, stream>>>(W_out, Wobf);
  k_f2bf<<<dim3(wgrid * 16), 256, 0, stream>>>(x, xbf); // 16M elements
  k_decay<<<dim3(NDST / 256), 256, 0, stream>>>(log_decay, cdec);

  k_gates<<<dim3(NM / 64, NDST / 128), 256, 0, stream>>>(
      xbf, Wbf, b_in, b_r, b_i, cdec, a_ws, u_ws);
  k_scan<<<dim3((NB * NDST) / 256), 256, 0, stream>>>(a_ws, u_ws, h0, hs,
                                                      h_last);
  k_out<<<dim3(NM / 64, NDIN / 256), 256, 0, stream>>>(hs, Wobf, b_out, out);
}
